// RGA_Module_C_89567247991419
// MI455X (gfx1250) — matrix operand order repaired and checked by reading, not compiled, not run on hardware
//
#include <hip/hip_runtime.h>

// ---------- vector types (trivial, union-safe) ----------
typedef __attribute__((ext_vector_type(16))) __bf16         v16bf;
typedef __attribute__((ext_vector_type(8)))  float          v8f;
typedef __attribute__((ext_vector_type(16))) unsigned short u16x16;
typedef __attribute__((ext_vector_type(4)))  unsigned int   u32x4;
typedef __attribute__((ext_vector_type(4)))  float          f32x4;

#define BM 128
#define BN 128
#define KT 32
#define KTP 40                            // padded LDS row stride (bf16 elems)
#define TILE_USH (128 * KTP)              // 5120 ushorts per tile
#define BUF_USH  (2 * TILE_USH)           // A+B tiles per buffer
#define BUF_BYTES (BUF_USH * 2)           // 20480 bytes
#define NBUF 3                            // triple buffer, 2-deep prefetch

__device__ __forceinline__ unsigned short f2bf(float f) {
    unsigned u = __float_as_uint(f);
    unsigned r = (u + 0x7FFFu + ((u >> 16) & 1u)) >> 16;   // RNE
    return (unsigned short)r;
}

__device__ __forceinline__ float bnrelu(float v, const float* __restrict__ bn,
                                        int C, int ch, bool relu) {
    float g = bn[ch], b = bn[C + ch], m = bn[2 * C + ch], var = bn[3 * C + ch];
    float o = g * (v - m) * rsqrtf(var + 1e-5f) + b;
    return relu ? fmaxf(o, 0.f) : o;
}

// CDNA5 async copy: global -> LDS, 16B per active lane, tracked by ASYNCcnt.
__device__ __forceinline__ void async16(unsigned ldsoff, unsigned long long gaddr) {
    asm volatile("global_load_async_to_lds_b128 %0, %1, off"
                 :: "v"(ldsoff), "v"(gaddr) : "memory");
}
template <int N>
__device__ __forceinline__ void wait_async_le() {
    asm volatile("s_wait_asynccnt %0" :: "n"(N) : "memory");
}

// ---------------------------------------------------------------------------
// pack_weights: Wstack = bf16([W_theta; W_phi; W_gx])  [1728 x 2304]
//               WggB   = bf16(W_gg)                    [128 x 1024]
// ---------------------------------------------------------------------------
__global__ void pack_weights(const float* __restrict__ Wt, const float* __restrict__ Wp,
                             const float* __restrict__ Wg, const float* __restrict__ Wgg,
                             unsigned short* __restrict__ Wstack,
                             unsigned short* __restrict__ WggB) {
    long i = (long)blockIdx.x * 256 + threadIdx.x;
    const long NS = 1728L * 2304L;
    if (i < NS) {
        int m = (int)(i / 2304), k = (int)(i % 2304);
        float v = (m < 576)  ? Wt[(long)m * 2304 + k]
                : (m < 1152) ? Wp[(long)(m - 576) * 2304 + k]
                             : Wg[(long)(m - 1152) * 2304 + k];
        Wstack[i] = f2bf(v);
    } else if (i < NS + 131072L) {
        long j = i - NS;
        WggB[j] = f2bf(Wgg[j]);
    }
}

// convert_x: x fp32 [b,512,2304] -> bf16 same layout (8 elems / thread)
__global__ __launch_bounds__(256)
void convert_x(const float* __restrict__ x, unsigned short* __restrict__ xb, long total8) {
    long i = (long)blockIdx.x * 256 + threadIdx.x;
    if (i >= total8) return;
    const f32x4* src = (const f32x4*)x;
    f32x4 a = src[2 * i], c = src[2 * i + 1];
    union { unsigned short s[8]; u32x4 q; } u;
#pragma unroll
    for (int e = 0; e < 4; e++) { u.s[e] = f2bf(a[e]); u.s[4 + e] = f2bf(c[e]); }
    ((u32x4*)xb)[i] = u.q;
}

// ---------------------------------------------------------------------------
// bf16 WMMA GEMM:  C[M,N] = A[M,K] * B[K,N]
//   A : bf16 row-major [M,K], batch stride aStride (0 = shared)
//   B : passed transposed, bf16 row-major [N,K]
// 256 threads = 8 waves (2Mx4N), wave tile 64x32, K-step 32.
// Triple-buffered LDS, 2-deep global_load_async_to_lds_b128 prefetch;
// every wave issues exactly 4 asyncs/slab (A rows clamped to M-1), so
// s_wait_asynccnt 4 == "previous slab landed, newest still in flight".
// MODE 0: stacked theta/phi/gx epilogue.  MODE 1: Gc both orientations.
// MODE 2: bn_gg+relu -> f32.
// ---------------------------------------------------------------------------
template <int MODE>
__global__ __launch_bounds__(256)
void gemm_bf16(const unsigned short* __restrict__ A, long aStride,
               const unsigned short* __restrict__ B, long bStride,
               int M, int N, int K,
               const float* __restrict__ bn0, const float* __restrict__ bn1,
               const float* __restrict__ bn2,
               unsigned short* __restrict__ out0, long o0Stride,
               unsigned short* __restrict__ out1, long o1Stride,
               float* __restrict__ outf, long ofStride) {
    __shared__ __align__(16) unsigned short SMEM[NBUF * BUF_USH];

    const int b    = blockIdx.z;
    const int m0   = blockIdx.y * BM;
    const int n0   = blockIdx.x * BN;
    const int t    = threadIdx.x;
    const int lane = t & 31;
    const int wave = t >> 5;
    const int mw = wave >> 2, nw = wave & 3;
    const int lh = lane >> 4, l15 = lane & 15;

    // staging assignment: row r, 32B half hf (two 16B async loads each).
    // A row clamped to M-1: uniform 4 asyncs/slab per wave; clamped rows only
    // feed output rows mg>=M which the epilogue never stores (M % 16 == 0).
    const int r  = t >> 1;
    const int hf = t & 1;
    const int ar = (m0 + r < M) ? (m0 + r) : (M - 1);
    const unsigned long long gA =
        (unsigned long long)(A + (long)b * aStride + (long)ar * K + hf * 16);
    const unsigned long long gB =
        (unsigned long long)(B + (long)b * bStride + (long)(n0 + r) * K + hf * 16);
    // generic LDS pointer: addr[31:0] is the LDS byte offset
    const unsigned ldsA = (unsigned)(size_t)&SMEM[r * KTP + hf * 16];
    const unsigned ldsB = ldsA + TILE_USH * 2;

    v8f acc[4][2];
#pragma unroll
    for (int i = 0; i < 4; i++)
#pragma unroll
        for (int j = 0; j < 2; j++)
            acc[i][j] = (v8f){0.f, 0.f, 0.f, 0.f, 0.f, 0.f, 0.f, 0.f};

    const int nk = K / KT;

    auto issue_slab = [&](int ks, int buf) {
        const unsigned long long go = 64ull * (unsigned long long)ks;  // KT bf16 = 64B
        const unsigned bo = (unsigned)(buf * BUF_BYTES);
        async16(ldsA + bo, gA + go); async16(ldsA + bo + 16, gA + go + 16);
        async16(ldsB + bo, gB + go); async16(ldsB + bo + 16, gB + go + 16);
    };

    // prologue: fill buffers 0 and 1
    issue_slab(0, 0);
    if (nk > 1) { issue_slab(1, 1); wait_async_le<4>(); }
    else        { wait_async_le<0>(); }
    __syncthreads();

    int cur = 0;
    for (int ks = 0; ks < nk; ks++) {
        const bool havePf = (ks + 2 < nk);
        if (havePf) {
            int pf = cur + 2; if (pf >= NBUF) pf -= NBUF;
            issue_slab(ks + 2, pf);
        }

        const unsigned short* As = SMEM + cur * BUF_USH;
        const unsigned short* Bs = As + TILE_USH;

        // A frag: lane half lh -> K {lh*8..+7, lh*8+16..+23}; row = l15
        u16x16 afr[4], bfr[2];
        union { u32x4 q[2]; u16x16 v; } u;
#pragma unroll
        for (int mf = 0; mf < 4; mf++) {
            int row = mw * 64 + mf * 16 + l15;
            u.q[0] = *(const u32x4*)&As[row * KTP + lh * 8];
            u.q[1] = *(const u32x4*)&As[row * KTP + lh * 8 + 16];
            afr[mf] = u.v;
        }
        // B frag: col = l15, lane half lh -> K {lh*8..+7, lh*8+16..+23}
#pragma unroll
        for (int nf = 0; nf < 2; nf++) {
            int nrow = nw * 32 + nf * 16 + l15;
            const u32x4* q = (const u32x4*)&Bs[nrow * KTP + lh * 8];
            u.q[0] = q[0]; u.q[1] = q[2];
            bfr[nf] = u.v;
        }
#pragma unroll
        for (int mf = 0; mf < 4; mf++)
#pragma unroll
            for (int nf = 0; nf < 2; nf++)
                acc[mf][nf] = __builtin_amdgcn_wmma_f32_16x16x32_bf16(
                    false, __builtin_bit_cast(v16bf, afr[mf]),
                    false, __builtin_bit_cast(v16bf, bfr[nf]),
                    (short)0, acc[mf][nf], false, false);

        if (ks + 1 < nk) {
            if (havePf) wait_async_le<4>();  // slab ks+1 landed; ks+2 in flight
            else        wait_async_le<0>();  // drain tail
            __syncthreads();
        }
        cur = cur + 1 == NBUF ? 0 : cur + 1;
    }

    // ---- epilogue ----
    float gaccum[2] = {0.f, 0.f};
#pragma unroll
    for (int mf = 0; mf < 4; mf++) {
#pragma unroll
        for (int nf = 0; nf < 2; nf++) {
#pragma unroll
            for (int rr = 0; rr < 8; rr++) {
                int mg = m0 + mw * 64 + mf * 16 + lh * 8 + rr;
                int ng = n0 + nw * 32 + nf * 16 + l15;
                float val = acc[mf][nf][rr];
                if (MODE == 0) {
                    if (mg < 576) {                 // theta -> xc1T [512 x 576]
                        float o = bnrelu(val, bn0, 576, mg, true);
                        out0[(long)b * o0Stride + (long)ng * 576 + mg] = f2bf(o);
                    } else if (mg < 1152) {         // phi -> phiT [512 x 576]
                        float o = bnrelu(val, bn1, 576, mg - 576, true);
                        out1[(long)b * o1Stride + (long)ng * 576 + (mg - 576)] = f2bf(o);
                    } else if (mg < 1728) {         // gx -> mean accumulation
                        float o = bnrelu(val, bn2, 576, mg - 1152, true);
                        gaccum[nf] += o * (1.0f / 576.0f);
                    }
                } else if (MODE == 1) {             // Gc both orientations
                    unsigned short hv = f2bf(val);
                    out0[(long)b * o0Stride + (long)mg * 1024 + ng] = hv;
                    out0[(long)b * o0Stride + (long)ng * 1024 + 512 + mg] = hv;
                } else {                            // MODE 2: bn_gg + relu
                    float o = bnrelu(val, bn0, 128, mg, true);
                    outf[(long)b * ofStride + (long)mg * 512 + ng] = o;
                }
            }
        }
    }
    if (MODE == 0) {
#pragma unroll
        for (int nf = 0; nf < 2; nf++) {
            if (gaccum[nf] != 0.f) {
                int ng = n0 + nw * 32 + nf * 16 + l15;
                atomicAdd(&outf[(long)b * ofStride + ng], gaccum[nf]);
            }
        }
    }
}

// ---------------------------------------------------------------------------
// finalize: per (b, n): yc = [g ; GJout[:,n]] (129) -> CBR(W_c1) -> BN(W_c2)
//           -> sigmoid -> gate[b,n]
// ---------------------------------------------------------------------------
__global__ __launch_bounds__(512)
void finalize_kernel(const float* __restrict__ gsum, const float* __restrict__ GJout,
                     const float* __restrict__ Wc1, const float* __restrict__ bnc1,
                     const float* __restrict__ Wc2, const float* __restrict__ bnc2,
                     float* __restrict__ gate) {
    __shared__ float sW[64 * 129];
    __shared__ float sb[4 * 64 + 64 + 4];
    int b = blockIdx.x, t = threadIdx.x;
    for (int i = t; i < 64 * 129; i += 512) sW[i] = Wc1[i];
    for (int i = t; i < 256; i += 512) sb[i] = bnc1[i];
    if (t < 64) sb[256 + t] = Wc2[t];
    if (t < 4)  sb[320 + t] = bnc2[t];
    __syncthreads();

    int n = t;
    float g = gsum[b * 512 + n];
    const float* col = GJout + (long)b * 128 * 512 + n;
    float acc2 = 0.f;
    for (int k = 0; k < 64; k++) {
        float s = sW[k * 129] * g;
#pragma unroll 4
        for (int j = 0; j < 128; j++) s += sW[k * 129 + 1 + j] * col[(long)j * 512];
        float o = sb[k] * (s - sb[128 + k]) * rsqrtf(sb[192 + k] + 1e-5f) + sb[64 + k];
        o = fmaxf(o, 0.f);
        acc2 += sb[256 + k] * o;
    }
    float wy = sb[320] * (acc2 - sb[322]) * rsqrtf(sb[323] + 1e-5f) + sb[321];
    gate[b * 512 + n] = 1.f / (1.f + __expf(-wy));
}

// ---------------------------------------------------------------------------
// apply_gate: out = x * (1 + sigmoid(wy))   (float4, pure bandwidth)
// ---------------------------------------------------------------------------
__global__ __launch_bounds__(256)
void apply_gate_kernel(const float* __restrict__ x, const float* __restrict__ gate,
                       float* __restrict__ out, long total4) {
    long i = (long)blockIdx.x * 256 + threadIdx.x;
    if (i >= total4) return;
    const f32x4* xi = (const f32x4*)x;
    f32x4* oi = (f32x4*)out;
    f32x4 v = xi[i];
    int bc = (int)((i * 4) / 2304);       // hw = 2304, divisible by 4
    float s = 1.f + gate[bc];
    oi[i] = v * s;
}

// ---------------------------------------------------------------------------
extern "C" void kernel_launch(void* const* d_in, const int* in_sizes, int n_in,
                              void* d_out, int out_size, void* d_ws, size_t ws_size,
                              hipStream_t stream) {
    const float* x     = (const float*)d_in[0];
    const float* Wth   = (const float*)d_in[1];
    const float* bn_th = (const float*)d_in[2];
    const float* Wph   = (const float*)d_in[3];
    const float* bn_ph = (const float*)d_in[4];
    const float* Wgx   = (const float*)d_in[5];
    const float* bn_gx = (const float*)d_in[6];
    const float* Wgg   = (const float*)d_in[7];
    const float* bn_gg = (const float*)d_in[8];
    const float* Wc1   = (const float*)d_in[9];
    const float* bn_c1 = (const float*)d_in[10];
    const float* Wc2   = (const float*)d_in[11];
    const float* bn_c2 = (const float*)d_in[12];
    float* out = (float*)d_out;

    char* ws = (char*)d_ws;
    auto carve = [&](size_t bytes) -> char* {
        char* p = ws;
        ws += (bytes + 255) & ~(size_t)255;
        return p;
    };
    unsigned short* xbf    = (unsigned short*)carve(32UL * 512 * 2304 * 2); // 75.5 MB
    unsigned short* Wstack = (unsigned short*)carve(1728UL * 2304 * 2);     // 8.0 MB
    unsigned short* WggB   = (unsigned short*)carve(128UL * 1024 * 2);      // 0.26 MB
    unsigned short* xc1T   = (unsigned short*)carve(32UL * 512 * 576 * 2);  // 18.9 MB
    unsigned short* phiT   = (unsigned short*)carve(32UL * 512 * 576 * 2);  // 18.9 MB
    unsigned short* GJT    = (unsigned short*)carve(32UL * 512 * 1024 * 2); // 33.6 MB
    float*          GJout  = (float*)carve(32UL * 128 * 512 * 4);           // 8.4 MB
    float*          gsum   = (float*)carve(32UL * 512 * 4);
    float*          gate   = (float*)carve(32UL * 512 * 4);

    pack_weights<<<16064, 256, 0, stream>>>(Wth, Wph, Wgx, Wgg, Wstack, WggB);
    convert_x<<<18432, 256, 0, stream>>>(x, xbf, 4718592L);
    hipMemsetAsync(gsum, 0, 32UL * 512 * 4, stream);

    // fused theta/phi/gx GEMM: [1728x2304] * x[b]^T   (Bt = xbf [512 x 2304])
    gemm_bf16<0><<<dim3(4, 14, 32), 256, 0, stream>>>(
        Wstack, 0, xbf, 512L * 2304, 1728, 512, 2304,
        bn_th, bn_ph, bn_gx,
        xc1T, 512L * 576, phiT, 512L * 576, gsum, 512);

    // Gc = thetaT * phi : A=xc1T [512x576], Bt=phiT [512x576]
    gemm_bf16<1><<<dim3(4, 4, 32), 256, 0, stream>>>(
        xc1T, 512L * 576, phiT, 512L * 576, 512, 512, 576,
        nullptr, nullptr, nullptr,
        GJT, 512L * 1024, nullptr, 0, nullptr, 0);

    // W_gg * [Gc^T ; Gc] : A=WggB [128x1024], Bt=GJT [512x1024]
    gemm_bf16<2><<<dim3(4, 1, 32), 256, 0, stream>>>(
        WggB, 0, GJT, 512L * 1024, 128, 512, 1024,
        bn_gg, nullptr, nullptr,
        nullptr, 0, nullptr, 0, GJout, 128L * 512);

    finalize_kernel<<<32, 512, 0, stream>>>(gsum, GJout, Wc1, bn_c1, Wc2, bn_c2, gate);

    apply_gate_kernel<<<36864, 256, 0, stream>>>(x, gate, out, 9437184L);
}
